// E67HGated_4887672783132
// MI455X (gfx1250) — matrix operand order repaired and checked by reading, not compiled, not run on hardware
//
#include <hip/hip_runtime.h>

// ---------------------------------------------------------------------------
// Types for CDNA5 WMMA (wave32): v16bf A/B fragments, v8f accumulators.
// ---------------------------------------------------------------------------
typedef __bf16 bf16_t;
typedef __attribute__((ext_vector_type(16))) __bf16 v16bf;
typedef __attribute__((ext_vector_type(8)))  __bf16 v8bf;
typedef __attribute__((ext_vector_type(8)))  float  v8f;

#define DEV __device__ __forceinline__

DEV float bf2f(bf16_t b) {
  unsigned short s = __builtin_bit_cast(unsigned short, b);
  unsigned u = ((unsigned)s) << 16;
  return __builtin_bit_cast(float, u);
}
DEV bf16_t f2bf(float f) {
  unsigned u = __builtin_bit_cast(unsigned, f);
  unsigned r = u + 0x7FFFu + ((u >> 16) & 1u);   // round-to-nearest-even
  unsigned short s = (unsigned short)(r >> 16);
  return __builtin_bit_cast(bf16_t, s);
}
DEV unsigned short f2bf_bits(float f) {
  unsigned u = __builtin_bit_cast(unsigned, f);
  unsigned r = u + 0x7FFFu + ((u >> 16) & 1u);
  return (unsigned short)(r >> 16);
}
// Fast sigmoid/tanh: v_exp_f32 + v_rcp_f32 (keeps IEEE div sequences off the
// critical path; precision ~1ulp rcp is far above bf16 storage precision).
DEV float sigmoidf_(float x) {
  return __builtin_amdgcn_rcpf(1.0f + __expf(-x));
}
DEV float siluf_(float x) { return x * sigmoidf_(x); }
DEV float tanhf_(float x) {
  return 1.0f - 2.0f * __builtin_amdgcn_rcpf(__expf(2.0f * x) + 1.0f);
}

union FragU { v16bf v; v8bf h[2]; };

// A-fragment 16x32 bf16, row-major source, per ISA layout:
// lanes 0-15: V0-3 = K0..7, V4-7 = K16..23 ; lanes 16-31: K8..15 / K24..31
DEV v16bf load_frag_a(const bf16_t* __restrict__ A, int lda, int lane) {
  const int m = lane & 15, half = lane >> 4;
  const bf16_t* p = A + (size_t)m * lda + half * 8;
  FragU f;
  f.h[0] = *(const v8bf*)(p);
  f.h[1] = *(const v8bf*)(p + 16);
  return f.v;
}

// B-fragment 32x16 bf16 built from row-major W[N,K] (we compute A * W^T):
// lane n holds column n of B == row n of W; lanes 0-15 K0..7/K16..23, 16-31 K8..15/K24..31
DEV v16bf load_frag_b(const bf16_t* __restrict__ W, int ldw, int lane) {
  const int n = lane & 15, half = lane >> 4;
  const bf16_t* p = W + (size_t)n * ldw + half * 8;
  FragU f;
  f.h[0] = *(const v8bf*)(p);
  f.h[1] = *(const v8bf*)(p + 16);
  return f.v;
}

DEV v8f wmma_bf16(v16bf a, v16bf b, v8f c) {
  return __builtin_amdgcn_wmma_f32_16x16x32_bf16(
      /*neg_a=*/false, a, /*neg_b=*/false, b,
      /*c_mod=*/(short)0, c, /*reuse_a=*/false, /*reuse_b=*/false);
}

// ---------------------------------------------------------------------------
// f32 -> bf16 conversion, vectorized 4-wide (float4 in, 4x bf16 = 8B out)
// ---------------------------------------------------------------------------
__global__ __launch_bounds__(256) void cvt_f32_bf16_v4(
    const float4* __restrict__ src, ushort4* __restrict__ dst, long n4) {
  long i = (long)blockIdx.x * blockDim.x + threadIdx.x;
  long stride = (long)gridDim.x * blockDim.x;
  for (; i < n4; i += stride) {
    float4 f = src[i];
    ushort4 o;
    o.x = f2bf_bits(f.x);
    o.y = f2bf_bits(f.y);
    o.z = f2bf_bits(f.z);
    o.w = f2bf_bits(f.w);
    dst[i] = o;
  }
}

// ---------------------------------------------------------------------------
// GEMM1: C = silu(A * W^T), bf16 out. Wave tile 64x64, block tile 256x128.
// ---------------------------------------------------------------------------
constexpr int G1_WM = 4, G1_WN = 4, G1_WAVES_M = 4, G1_WAVES_N = 2;
constexpr int G1_BM = G1_WAVES_M * G1_WM * 16;  // 256
constexpr int G1_BN = G1_WAVES_N * G1_WN * 16;  // 128

__global__ __launch_bounds__(256) void gemm_silu_kernel(
    const bf16_t* __restrict__ A, const bf16_t* __restrict__ W,
    bf16_t* __restrict__ C, int Ndim, int Kdim) {
  const int lane = threadIdx.x & 31;
  const int wave = threadIdx.x >> 5;
  const int wm = wave & (G1_WAVES_M - 1);
  const int wn = wave / G1_WAVES_M;
  const int m0 = blockIdx.y * G1_BM + wm * (G1_WM * 16);
  const int n0 = blockIdx.x * G1_BN + wn * (G1_WN * 16);

  v8f acc[G1_WM][G1_WN];
#pragma unroll
  for (int i = 0; i < G1_WM; ++i)
#pragma unroll
    for (int j = 0; j < G1_WN; ++j)
#pragma unroll
      for (int e = 0; e < 8; ++e) acc[i][j][e] = 0.0f;

  for (int k0 = 0; k0 < Kdim; k0 += 32) {
    v16bf a[G1_WM], b[G1_WN];
#pragma unroll
    for (int i = 0; i < G1_WM; ++i)
      a[i] = load_frag_a(A + (size_t)(m0 + i * 16) * Kdim + k0, Kdim, lane);
#pragma unroll
    for (int j = 0; j < G1_WN; ++j)
      b[j] = load_frag_b(W + (size_t)(n0 + j * 16) * Kdim + k0, Kdim, lane);
#pragma unroll
    for (int i = 0; i < G1_WM; ++i)
#pragma unroll
      for (int j = 0; j < G1_WN; ++j)
        acc[i][j] = wmma_bf16(a[i], b[j], acc[i][j]);
  }

  const int col = lane & 15;
  const int rbase = (lane >> 4) * 8;
#pragma unroll
  for (int i = 0; i < G1_WM; ++i)
#pragma unroll
    for (int j = 0; j < G1_WN; ++j)
#pragma unroll
      for (int r = 0; r < 8; ++r) {
        int row = m0 + i * 16 + rbase + r;
        C[(size_t)row * Ndim + n0 + j * 16 + col] = f2bf(siluf_(acc[i][j][r]));
      }
}

// ---------------------------------------------------------------------------
// GEMM2 (fused dual): AX = A*Wa^T (raw, bf16), V = tanh(A*Wx^T + b_v) (bf16).
// Shares A fragments across both weight streams. Wave tile 64x32, block 256x64.
// ---------------------------------------------------------------------------
constexpr int G2_WM = 4, G2_WN = 2, G2_WAVES_M = 4, G2_WAVES_N = 2;
constexpr int G2_BM = G2_WAVES_M * G2_WM * 16;  // 256
constexpr int G2_BN = G2_WAVES_N * G2_WN * 16;  // 64

__global__ __launch_bounds__(256) void gemm_dual_kernel(
    const bf16_t* __restrict__ A,
    const bf16_t* __restrict__ Wa, const bf16_t* __restrict__ Wx,
    const float* __restrict__ bv,
    bf16_t* __restrict__ AX, bf16_t* __restrict__ V,
    int Ndim, int Kdim) {
  const int lane = threadIdx.x & 31;
  const int wave = threadIdx.x >> 5;
  const int wm = wave & (G2_WAVES_M - 1);
  const int wn = wave / G2_WAVES_M;
  const int m0 = blockIdx.y * G2_BM + wm * (G2_WM * 16);
  const int n0 = blockIdx.x * G2_BN + wn * (G2_WN * 16);

  v8f acc_a[G2_WM][G2_WN], acc_v[G2_WM][G2_WN];
#pragma unroll
  for (int i = 0; i < G2_WM; ++i)
#pragma unroll
    for (int j = 0; j < G2_WN; ++j)
#pragma unroll
      for (int e = 0; e < 8; ++e) { acc_a[i][j][e] = 0.0f; acc_v[i][j][e] = 0.0f; }

  for (int k0 = 0; k0 < Kdim; k0 += 32) {
    v16bf a[G2_WM], ba[G2_WN], bx[G2_WN];
#pragma unroll
    for (int i = 0; i < G2_WM; ++i)
      a[i] = load_frag_a(A + (size_t)(m0 + i * 16) * Kdim + k0, Kdim, lane);
#pragma unroll
    for (int j = 0; j < G2_WN; ++j) {
      ba[j] = load_frag_b(Wa + (size_t)(n0 + j * 16) * Kdim + k0, Kdim, lane);
      bx[j] = load_frag_b(Wx + (size_t)(n0 + j * 16) * Kdim + k0, Kdim, lane);
    }
#pragma unroll
    for (int i = 0; i < G2_WM; ++i)
#pragma unroll
      for (int j = 0; j < G2_WN; ++j) {
        acc_a[i][j] = wmma_bf16(a[i], ba[j], acc_a[i][j]);
        acc_v[i][j] = wmma_bf16(a[i], bx[j], acc_v[i][j]);
      }
  }

  const int col = lane & 15;
  const int rbase = (lane >> 4) * 8;
#pragma unroll
  for (int j = 0; j < G2_WN; ++j) {
    const float bvv = bv[n0 + j * 16 + col];
#pragma unroll
    for (int i = 0; i < G2_WM; ++i)
#pragma unroll
      for (int r = 0; r < 8; ++r) {
        int row = m0 + i * 16 + rbase + r;
        size_t o = (size_t)row * Ndim + n0 + j * 16 + col;
        AX[o] = f2bf(acc_a[i][j][r]);
        V[o]  = f2bf(tanhf_(acc_v[i][j][r] + bvv));
      }
  }
}

// ---------------------------------------------------------------------------
// Sequential gated scan over T. One thread per (b,d) channel; f32 state.
// Critical path per step is fma -> exp -> rcp -> fma (everything else,
// including the ax+b_alpha add and the h*silu(h) output, is off-chain).
// ---------------------------------------------------------------------------
__global__ __launch_bounds__(256) void scan_kernel(
    const bf16_t* __restrict__ ax, const bf16_t* __restrict__ v,
    const float* __restrict__ h0,
    const float* __restrict__ d_alpha, const float* __restrict__ b_alpha,
    bf16_t* __restrict__ cell, float* __restrict__ h_final,
    int Bn, int Tn, int Dn) {
  int idx = blockIdx.x * blockDim.x + threadIdx.x;
  if (idx >= Bn * Dn) return;
  const int b = idx / Dn;
  const int d = idx - b * Dn;

  float h = h0[idx];
  const float da = d_alpha[d];
  const float ba = b_alpha[d];

  const size_t base = (size_t)b * Tn * Dn + d;
  const bf16_t* axp = ax + base;
  const bf16_t* vp  = v + base;
  bf16_t* cp        = cell + base;

#pragma unroll 4
  for (int t = 0; t < Tn; ++t) {
    const size_t o = (size_t)t * Dn;
    float axb = bf2f(axp[o]) + ba;          // off-chain
    float vv  = bf2f(vp[o]);                // off-chain
    // chain: fma -> exp -> rcp -> fma
    float z     = da * h + axb;
    float alpha = __builtin_amdgcn_rcpf(1.0f + __expf(-z));
    h = vv + alpha * (h - vv);
    // output path, off the recurrence chain
    float sig = __builtin_amdgcn_rcpf(1.0f + __expf(-h));
    cp[o] = f2bf(h * h * sig);
  }
  h_final[idx] = h;
}

// ---------------------------------------------------------------------------
// GEMM3: OUT = cell * W_out^T, f32 out straight into d_out.
// ---------------------------------------------------------------------------
__global__ __launch_bounds__(256) void gemm_f32out_kernel(
    const bf16_t* __restrict__ A, const bf16_t* __restrict__ W,
    float* __restrict__ C, int Ndim, int Kdim) {
  const int lane = threadIdx.x & 31;
  const int wave = threadIdx.x >> 5;
  const int wm = wave & (G1_WAVES_M - 1);
  const int wn = wave / G1_WAVES_M;
  const int m0 = blockIdx.y * G1_BM + wm * (G1_WM * 16);
  const int n0 = blockIdx.x * G1_BN + wn * (G1_WN * 16);

  v8f acc[G1_WM][G1_WN];
#pragma unroll
  for (int i = 0; i < G1_WM; ++i)
#pragma unroll
    for (int j = 0; j < G1_WN; ++j)
#pragma unroll
      for (int e = 0; e < 8; ++e) acc[i][j][e] = 0.0f;

  for (int k0 = 0; k0 < Kdim; k0 += 32) {
    v16bf a[G1_WM], b[G1_WN];
#pragma unroll
    for (int i = 0; i < G1_WM; ++i)
      a[i] = load_frag_a(A + (size_t)(m0 + i * 16) * Kdim + k0, Kdim, lane);
#pragma unroll
    for (int j = 0; j < G1_WN; ++j)
      b[j] = load_frag_b(W + (size_t)(n0 + j * 16) * Kdim + k0, Kdim, lane);
#pragma unroll
    for (int i = 0; i < G1_WM; ++i)
#pragma unroll
      for (int j = 0; j < G1_WN; ++j)
        acc[i][j] = wmma_bf16(a[i], b[j], acc[i][j]);
  }

  const int col = lane & 15;
  const int rbase = (lane >> 4) * 8;
#pragma unroll
  for (int i = 0; i < G1_WM; ++i)
#pragma unroll
    for (int j = 0; j < G1_WN; ++j)
#pragma unroll
      for (int r = 0; r < 8; ++r) {
        int row = m0 + i * 16 + rbase + r;
        C[(size_t)row * Ndim + n0 + j * 16 + col] = acc[i][j][r];
      }
}

// ---------------------------------------------------------------------------
// Host-side launcher
// ---------------------------------------------------------------------------
extern "C" void kernel_launch(void* const* d_in, const int* in_sizes, int n_in,
                              void* d_out, int out_size, void* d_ws, size_t ws_size,
                              hipStream_t stream) {
  (void)in_sizes; (void)n_in; (void)out_size; (void)ws_size;
  const int Bn = 8, Tn = 2048, Dn = 1024;
  const int Mdim = Bn * Tn;  // 16384

  const float* x       = (const float*)d_in[0];
  const float* h0      = (const float*)d_in[1];
  const float* W_in    = (const float*)d_in[2];
  const float* W_alpha = (const float*)d_in[3];
  const float* d_alpha = (const float*)d_in[4];
  const float* b_alpha = (const float*)d_in[5];
  const float* W_x     = (const float*)d_in[6];
  const float* b_v     = (const float*)d_in[7];
  const float* W_out   = (const float*)d_in[8];

  float* outp  = (float*)d_out;              // [B,T,D]
  float* h_fin = outp + (size_t)Mdim * Dn;   // [B,D]

  char* ws = (char*)d_ws;
  size_t off = 0;
  auto take = [&](size_t bytes) -> char* {
    char* p = ws + off;
    off = (off + bytes + 255) & ~(size_t)255;
    return p;
  };
  bf16_t* x_bf    = (bf16_t*)take((size_t)Mdim * Dn * sizeof(bf16_t));
  bf16_t* win_bf  = (bf16_t*)take((size_t)Dn * Dn * sizeof(bf16_t));
  bf16_t* wal_bf  = (bf16_t*)take((size_t)Dn * Dn * sizeof(bf16_t));
  bf16_t* wx_bf   = (bf16_t*)take((size_t)Dn * Dn * sizeof(bf16_t));
  bf16_t* wout_bf = (bf16_t*)take((size_t)Dn * Dn * sizeof(bf16_t));
  bf16_t* xp_bf   = (bf16_t*)take((size_t)Mdim * Dn * sizeof(bf16_t));
  bf16_t* ax_bf   = (bf16_t*)take((size_t)Mdim * Dn * sizeof(bf16_t));
  bf16_t* v_bf    = (bf16_t*)take((size_t)Mdim * Dn * sizeof(bf16_t));
  bf16_t* cell_bf = (bf16_t*)take((size_t)Mdim * Dn * sizeof(bf16_t));

  // 1) convert activations + weights to bf16 (4-wide vectorized)
  cvt_f32_bf16_v4<<<1024, 256, 0, stream>>>(
      (const float4*)x, (ushort4*)x_bf, (long)Mdim * Dn / 4);
  cvt_f32_bf16_v4<<<256, 256, 0, stream>>>(
      (const float4*)W_in, (ushort4*)win_bf, (long)Dn * Dn / 4);
  cvt_f32_bf16_v4<<<256, 256, 0, stream>>>(
      (const float4*)W_alpha, (ushort4*)wal_bf, (long)Dn * Dn / 4);
  cvt_f32_bf16_v4<<<256, 256, 0, stream>>>(
      (const float4*)W_x, (ushort4*)wx_bf, (long)Dn * Dn / 4);
  cvt_f32_bf16_v4<<<256, 256, 0, stream>>>(
      (const float4*)W_out, (ushort4*)wout_bf, (long)Dn * Dn / 4);

  // 2) xp = silu(x @ W_in^T)
  {
    dim3 grid(Dn / G1_BN, Mdim / G1_BM);
    gemm_silu_kernel<<<grid, 256, 0, stream>>>(x_bf, win_bf, xp_bf, Dn, Dn);
  }

  // 3) ax = xp @ W_alpha^T ; v = tanh(xp @ W_x^T + b_v)   (fused)
  {
    dim3 grid(Dn / G2_BN, Mdim / G2_BM);
    gemm_dual_kernel<<<grid, 256, 0, stream>>>(xp_bf, wal_bf, wx_bf, b_v,
                                               ax_bf, v_bf, Dn, Dn);
  }

  // 4) sequential gated scan over T
  scan_kernel<<<(Bn * Dn) / 256, 256, 0, stream>>>(ax_bf, v_bf, h0, d_alpha,
                                                   b_alpha, cell_bf, h_fin,
                                                   Bn, Tn, Dn);

  // 5) output = cell @ W_out^T  (f32, into d_out)
  {
    dim3 grid(Dn / G1_BN, Mdim / G1_BM);
    gemm_f32out_kernel<<<grid, 256, 0, stream>>>(cell_bf, wout_bf, outp, Dn, Dn);
  }
}
